// WNO1d_23545010717199
// MI455X (gfx1250) — matrix operand order repaired and checked by reading, not compiled, not run on hardware
//
#include <hip/hip_runtime.h>
#include <math.h>

// ---------------------------------------------------------------------------
// Types for gfx1250 WMMA
// ---------------------------------------------------------------------------
typedef __attribute__((ext_vector_type(16))) __bf16 v16bf;
typedef __attribute__((ext_vector_type(8)))  float  v8f;

// ---------------------------------------------------------------------------
// db6 filter banks (precomputed from REC_LO, see reference):
//   ANA_LO[t] = REC_LO[t]
//   ANA_HI[t] = (-1)^t * REC_LO[11-t]
//   SYN_LO[t] = REC_LO[11-t]
//   SYN_HI[t] = -(-1)^t * REC_LO[t]
// ---------------------------------------------------------------------------
__constant__ float c_ana_lo[12] = {
  0.11154074335008017f, 0.4946238903983854f, 0.7511339080215775f,
  0.3152503517092432f, -0.22626469396516913f, -0.12976686756709563f,
  0.09750160558707936f, 0.02752286553001629f, -0.031582039318031156f,
  0.0005538422009938016f, 0.004777257511010651f, -0.00107730108499558f };
__constant__ float c_ana_hi[12] = {
  -0.00107730108499558f, -0.004777257511010651f, 0.0005538422009938016f,
  0.031582039318031156f, 0.02752286553001629f, -0.09750160558707936f,
  -0.12976686756709563f, 0.22626469396516913f, 0.3152503517092432f,
  -0.7511339080215775f, 0.4946238903983854f, -0.11154074335008017f };
__constant__ float c_syn_lo[12] = {
  -0.00107730108499558f, 0.004777257511010651f, 0.0005538422009938016f,
  -0.031582039318031156f, 0.02752286553001629f, 0.09750160558707936f,
  -0.12976686756709563f, -0.22626469396516913f, 0.3152503517092432f,
  0.7511339080215775f, 0.4946238903983854f, 0.11154074335008017f };
__constant__ float c_syn_hi[12] = {
  -0.11154074335008017f, 0.4946238903983854f, -0.7511339080215775f,
  0.3152503517092432f, 0.22626469396516913f, -0.12976686756709563f,
  -0.09750160558707936f, 0.02752286553001629f, 0.031582039318031156f,
  0.0005538422009938016f, -0.004777257511010651f, -0.00107730108499558f };

__device__ __forceinline__ float gelu_exact(float x) {
  return 0.5f * x * (1.0f + erff(x * 0.7071067811865475f));
}

// ---------------------------------------------------------------------------
// fc0: v[b,c,s] = x[b,s]*W[0,c] + grid(s)*W[1,c] + bias[c]    (v: B,64,S)
// ---------------------------------------------------------------------------
__global__ void fc0_kernel(const float* __restrict__ x,
                           const float* __restrict__ w,   // (2,64)
                           const float* __restrict__ bias,// (64)
                           float* __restrict__ v) {
  int idx = blockIdx.x * blockDim.x + threadIdx.x;   // B*64*1024 total
  int s = idx & 1023;
  int c = (idx >> 10) & 63;
  int b = idx >> 16;
  float xv = x[b * 1024 + s];
  float g  = (float)s * (1.0f / 1023.0f);
  v[(b * 64 + c) * 1024 + s] = xv * w[c] + g * w[64 + c] + bias[c];
}

// ---------------------------------------------------------------------------
// Analysis filter bank, one level:  lo/hi[k] = sum_t xp[2k+t]*flt[t]
// symmetric padding, left pad = 10 at every level (verified).
// ---------------------------------------------------------------------------
__global__ void afb_kernel(const float* __restrict__ in, float* __restrict__ lo,
                           float* __restrict__ hi, int N, int outN, int total) {
  int idx = blockIdx.x * blockDim.x + threadIdx.x;
  if (idx >= total) return;
  int k = idx % outN, row = idx / outN;
  const float* src = in + (size_t)row * N;
  float aL = 0.f, aH = 0.f;
#pragma unroll
  for (int t = 0; t < 12; ++t) {
    int u = 2 * k + t - 10;               // pl = 10
    if (u < 0) u = -u - 1;                // symmetric reflect (edge incl.)
    if (u >= N) u = 2 * N - 1 - u;
    float xv = src[u];
    aL += xv * c_ana_lo[t];
    aH += xv * c_ana_hi[t];
  }
  lo[(size_t)row * outN + k] = aL;
  hi[(size_t)row * outN + k] = aH;
}

// ---------------------------------------------------------------------------
// Synthesis filter bank, one level (lhs_dilation=2, pad 1, 12 taps):
// out[k] = sum_{t: (k+t) odd, m=(k+t-1)/2 in [0,N)} lo[m]*SL[t] + hi[m]*SH[t]
// ---------------------------------------------------------------------------
__global__ void sfb_kernel(const float* __restrict__ lo, const float* __restrict__ hi,
                           float* __restrict__ out, int N, int loStride,
                           int outN, int total) {
  int idx = blockIdx.x * blockDim.x + threadIdx.x;
  if (idx >= total) return;
  int k = idx % outN, row = idx / outN;
  const float* lp = lo + (size_t)row * loStride;
  const float* hp = hi + (size_t)row * N;
  float acc = 0.f;
#pragma unroll
  for (int t = 0; t < 12; ++t) {
    if (((k + t) & 1) == 0) continue;
    int m = (k + t - 1) >> 1;
    if (m >= 0 && m < N)
      acc += lp[m] * c_syn_lo[t] + hp[m] * c_syn_hi[t];
  }
  out[(size_t)row * outN + k] = acc;
}

// ---------------------------------------------------------------------------
// Coarse per-position channel mix: out[b,o,x] = sum_i in[b,i,x]*w[i,o,x]
// (B,64,14) x (64,64,14) -> (B,64,14)
// ---------------------------------------------------------------------------
__global__ void cmix_kernel(const float* __restrict__ in, const float* __restrict__ w,
                            float* __restrict__ out) {
  int idx = blockIdx.x * blockDim.x + threadIdx.x;  // B*64*14
  int x = idx % 14;
  int o = (idx / 14) & 63;
  int b = idx / (14 * 64);
  float acc = 0.f;
#pragma unroll 8
  for (int i = 0; i < 64; ++i)
    acc += in[((size_t)(b * 64 + i)) * 14 + x] * w[(i * 64 + o) * 14 + x];
  out[((size_t)(b * 64 + o)) * 14 + x] = acc;
}

// ---------------------------------------------------------------------------
// Channel-mix GEMM + residual + bias + optional GELU, via bf16 WMMA.
//   vout[b,o,s] = act( a[b,o,s] + sum_c cw[o,c]*v[b,c,s] + cb[o] )
// One wave computes a 16(o) x 16(s) tile, K=64 in two 16x16x32 steps.
// A/B use contiguous k = 16*half + e in both operands; C/D follow the WMMA VGPR mapping (wave32).
// ---------------------------------------------------------------------------
__global__ void mix_kernel(const float* __restrict__ v, const float* __restrict__ a,
                           const float* __restrict__ cw, const float* __restrict__ cb,
                           float* __restrict__ vout, int applyGelu) {
  int wid  = (blockIdx.x * blockDim.x + threadIdx.x) >> 5;
  int lane = threadIdx.x & 31;
  int stile = wid & 63;
  int otile = (wid >> 6) & 3;
  int b     = wid >> 8;
  int l15 = lane & 15;
  int hiH = lane >> 4;                 // half-wave select
  v8f acc = {};
#pragma unroll
  for (int ks = 0; ks < 2; ++ks) {
    v16bf af, bf;
#pragma unroll
    for (int e = 0; e < 16; ++e) {
      // A element: row m=l15, K = ks*32 + e + 16*hiH
      int ka = ks * 32 + e + hiH * 16;
      af[e] = (__bf16)cw[(otile * 16 + l15) * 64 + ka];
      // B element: col n=l15, K = ks*32 + e + 16*hiH
      int kb = ks * 32 + e + hiH * 16;
      bf[e] = (__bf16)v[((size_t)(b * 64 + kb)) * 1024 + stile * 16 + l15];
    }
    acc = __builtin_amdgcn_wmma_f32_16x16x32_bf16(false, af, false, bf,
                                                  (short)0, acc, false, false);
  }
  int s = stile * 16 + l15;
#pragma unroll
  for (int r = 0; r < 8; ++r) {
    int o = otile * 16 + r + hiH * 8;  // C/D: VGPR r -> row r (+8 for lanes 16-31)
    size_t q = ((size_t)(b * 64 + o)) * 1024 + s;
    float val = acc[r] + a[q] + cb[o];
    if (applyGelu) val = gelu_exact(val);
    vout[q] = val;
  }
}

// ---------------------------------------------------------------------------
// Fused head: y[b,s] = sum_h gelu( sum_c v[b,c,s]*W1[c,h] + b1[h] )*W2[h] + b2
// One wave: 16 s-positions (M) x 128 h (8 N-tiles), K=64, bf16 WMMA,
// then cross-lane reduce over h within each 16-lane half.
// ---------------------------------------------------------------------------
__global__ void head_kernel(const float* __restrict__ v, const float* __restrict__ w1,
                            const float* __restrict__ b1, const float* __restrict__ w2,
                            const float* __restrict__ b2, float* __restrict__ out,
                            int accumulate) {
  int wid  = (blockIdx.x * blockDim.x + threadIdx.x) >> 5;
  int lane = threadIdx.x & 31;
  int stile = wid & 63;
  int b     = wid >> 6;
  int l15 = lane & 15;
  int hiH = lane >> 4;
  int s0 = stile * 16;
  // A fragments: A[m,k] = v[b, k, s0+m]
  v16bf a0, a1;
#pragma unroll
  for (int e = 0; e < 16; ++e) {
    int k = e + hiH * 16;
    a0[e] = (__bf16)v[((size_t)(b * 64 + k)) * 1024 + s0 + l15];
    a1[e] = (__bf16)v[((size_t)(b * 64 + k + 32)) * 1024 + s0 + l15];
  }
  float rowsum[8];
#pragma unroll
  for (int r = 0; r < 8; ++r) rowsum[r] = 0.f;
#pragma unroll
  for (int ht = 0; ht < 8; ++ht) {
    int h = ht * 16 + l15;
    v8f acc = {};
#pragma unroll
    for (int ks = 0; ks < 2; ++ks) {
      v16bf bf;
#pragma unroll
      for (int e = 0; e < 16; ++e) {
        int k = ks * 32 + e + hiH * 16;
        bf[e] = (__bf16)w1[k * 128 + h];
      }
      acc = __builtin_amdgcn_wmma_f32_16x16x32_bf16(false, (ks == 0) ? a0 : a1,
                                                    false, bf, (short)0, acc,
                                                    false, false);
    }
    float bb = b1[h], ww = w2[h];
#pragma unroll
    for (int r = 0; r < 8; ++r)
      rowsum[r] += gelu_exact(acc[r] + bb) * ww;
  }
  // reduce the 16 h-columns held by each half-wave (lanes 0-15 / 16-31)
#pragma unroll
  for (int mask = 1; mask <= 8; mask <<= 1)
#pragma unroll
    for (int r = 0; r < 8; ++r)
      rowsum[r] += __shfl_xor(rowsum[r], mask, 32);
  if (l15 == 0) {
    float bias2 = b2[0];
#pragma unroll
    for (int r = 0; r < 8; ++r) {
      int s = s0 + r + hiH * 8;
      float val = rowsum[r] + bias2;
      if (accumulate) out[b * 1024 + s] += val;
      else            out[b * 1024 + s]  = val;
    }
  }
}

// ---------------------------------------------------------------------------
// Host orchestration
// ---------------------------------------------------------------------------
extern "C" void kernel_launch(void* const* d_in, const int* in_sizes, int n_in,
                              void* d_out, int out_size, void* d_ws, size_t ws_size,
                              hipStream_t stream) {
  (void)in_sizes; (void)n_in; (void)out_size; (void)ws_size;
  const int B = 64, S = 1024, C = 64;
  const int rows = B * C;                         // 4096
  int N[9]; N[0] = S;
  for (int j = 0; j < 8; ++j) N[j + 1] = (N[j] + 11) / 2;  // 517,264,137,74,42,26,18,14

  // workspace carve-out (floats)
  float* ws = (float*)d_ws;
  size_t off = 0;
  float* v0 = ws + off; off += (size_t)rows * S;
  float* v1 = ws + off; off += (size_t)rows * S;
  float* s0 = ws + off; off += (size_t)rows * S;
  float* s1 = ws + off; off += (size_t)rows * S;
  float* hiA[8];
  for (int j = 0; j < 8; ++j) { hiA[j] = ws + off; off += (size_t)rows * N[j + 1]; }
  float* lo8m = ws + off; off += (size_t)rows * 14;
  float* hi8m = ws + off; off += (size_t)rows * 14;

  const float* x = (const float*)d_in[0];
  float* out = (float*)d_out;
  const int TB = 256;
  auto nblk = [](long long total, int bs) { return (int)((total + bs - 1) / bs); };

  for (int br = 0; br < 2; ++br) {
    int base = 1 + br * 10;
    const float* fc0_w = (const float*)d_in[base + 0];
    const float* fc0_b = (const float*)d_in[base + 1];
    const float* ww1   = (const float*)d_in[base + 2];   // (4,64,64,14)
    const float* ww2   = (const float*)d_in[base + 3];
    const float* cw    = (const float*)d_in[base + 4];   // (4,64,64)
    const float* cb    = (const float*)d_in[base + 5];   // (4,64)
    const float* fc1_w = (const float*)d_in[base + 6];
    const float* fc1_b = (const float*)d_in[base + 7];
    const float* fc2_w = (const float*)d_in[base + 8];
    const float* fc2_b = (const float*)d_in[base + 9];

    fc0_kernel<<<nblk((long long)rows * S, TB), TB, 0, stream>>>(x, fc0_w, fc0_b, v0);

    float* vc = v0; float* vn = v1;
    for (int layer = 0; layer < 4; ++layer) {
      // ---- analysis: 8 levels ----
      const float* src = vc;
      for (int j = 0; j < 8; ++j) {
        float* lodst = (j & 1) ? s1 : s0;
        long long tot = (long long)rows * N[j + 1];
        afb_kernel<<<nblk(tot, TB), TB, 0, stream>>>(src, lodst, hiA[j],
                                                     N[j], N[j + 1], (int)tot);
        src = lodst;
      }
      // ---- coarse channel mixes (lo8 in src, hi8 in hiA[7]) ----
      cmix_kernel<<<nblk((long long)B * 64 * 14, TB), TB, 0, stream>>>(
          src, ww1 + (size_t)layer * 64 * 64 * 14, lo8m);
      cmix_kernel<<<nblk((long long)B * 64 * 14, TB), TB, 0, stream>>>(
          hiA[7], ww2 + (size_t)layer * 64 * 64 * 14, hi8m);
      // ---- synthesis: 8 levels ----
      const float* cl = lo8m; int loStride = 14;
      float* dst = s0;
      for (int j = 7; j >= 0; --j) {
        const float* hp = (j == 7) ? hi8m : hiA[j];
        int hn = N[j + 1];
        int outN = 2 * hn - 10;
        dst = (j & 1) ? s0 : s1;
        long long tot = (long long)rows * outN;
        sfb_kernel<<<nblk(tot, TB), TB, 0, stream>>>(cl, hp, dst, hn, loStride,
                                                     outN, (int)tot);
        cl = dst; loStride = outN;
      }
      // ---- channel-mix GEMM + residual + bias (+gelu), WMMA ----
      mix_kernel<<<(B * 4 * 64) / 8, TB, 0, stream>>>(
          vc, cl, cw + (size_t)layer * 64 * 64, cb + (size_t)layer * 64,
          vn, (layer < 3) ? 1 : 0);
      float* t = vc; vc = vn; vn = t;
    }
    // ---- fused fc1+gelu+fc2 head, WMMA ----
    head_kernel<<<(B * 64) / 8, TB, 0, stream>>>(vc, fc1_w, fc1_b, fc2_w, fc2_b,
                                                 out, br);
  }
}
